// LowRankBilinearPooling_90563680403989
// MI455X (gfx1250) — matrix operand order repaired and checked by reading, not compiled, not run on hardware
//
#include <hip/hip_runtime.h>
#include <hip/hip_bf16.h>

// ---- problem constants -----------------------------------------------------
#define NB    16          // batch
#define NROW  196         // N1 == N2
#define CIN   768         // C1 == C2
#define HDIM  512         // H
#define ODIM  128         // O
#define KCHK  (CIN / 32)  // 24 K-chunks of 32
#define MTILE ((NROW + 15) / 16)  // 13 M-tiles (last one partial: 4 valid rows)

typedef __attribute__((ext_vector_type(16))) __bf16       v16bf;
typedef __attribute__((ext_vector_type(8)))  float        v8f;
typedef __attribute__((ext_vector_type(4)))  float        f32x4;
typedef __attribute__((ext_vector_type(4)))  unsigned int u32x4;

// ---- zero-init for the hsum accumulators (atomicAdd targets) --------------
__global__ void zero_f32_kernel(float* __restrict__ p, int n) {
    int i = blockIdx.x * blockDim.x + threadIdx.x;
    if (i < n) p[i] = 0.0f;
}

// ---- W [768,512] f32 -> bf16 swizzled as [kc][h][kr(0..31)] ----------------
// kr maps 1:1 to the per-lane B fragment: lane<16 reads kr 0..15 (K=kc*32+0..15),
// lane>=16 reads kr 16..31. Each lane's fragment is 32 contiguous bytes.
__global__ void swizzleW_kernel(const float* __restrict__ W, __bf16* __restrict__ Wb) {
    int idx = blockIdx.x * blockDim.x + threadIdx.x;   // over 768*512
    if (idx >= CIN * HDIM) return;
    int k  = idx >> 9;          // / 512
    int h  = idx & (HDIM - 1);  // % 512
    int kc = k >> 5;
    int kr = k & 31;
    Wb[(((size_t)kc * HDIM + h) << 5) + kr] = (__bf16)W[idx];
}

// ---- fused GEMM(bf16 WMMA, f32 acc) + relu + row-sum -----------------------
// grid = (13 M-tiles, 16 batches, 2 inputs), block = 256 (8 waves).
// Wave w owns columns [w*64, w*64+64) = 4 N-subtiles of 16.
__global__ __launch_bounds__(256)
void gemm_relu_colsum_kernel(const float* __restrict__ x1, const float* __restrict__ x2,
                             const __bf16* __restrict__ Wb1, const __bf16* __restrict__ Wb2,
                             float* __restrict__ h1s, float* __restrict__ h2s) {
    const int mt = blockIdx.x;
    const int b  = blockIdx.y;
    const int z  = blockIdx.z;

    const float*  __restrict__ x  = z ? x2  : x1;
    const __bf16* __restrict__ Wb = z ? Wb2 : Wb1;
    float*        __restrict__ hs = z ? h2s : h1s;

    const int wave  = threadIdx.x >> 5;
    const int lane  = threadIdx.x & 31;
    const int lhalf = lane >> 4;   // 0: K-low half, 1: K-high half
    const int l16   = lane & 15;

    const int  row      = mt * 16 + l16;            // A-matrix row this lane supplies
    const bool rowValid = (row < NROW);
    const float* __restrict__ arow =
        x + ((size_t)b * NROW + (rowValid ? row : 0)) * CIN;

    const int nbase = wave * 64;

    v8f acc0 = {}, acc1 = {}, acc2 = {}, acc3 = {};

    for (int kc = 0; kc < KCHK; ++kc) {
        // ---- A fragment: 16x32 bf16, contiguous K order, same as B ------------
        // lane<16 : e[0..15]=K{0..15}
        // lane>=16: e[0..15]=K{16..31}
        union { v16bf v; __bf16 e[16]; } A;
        if (rowValid) {
            const float* ap = arow + kc * 32 + lhalf * 16;
            f32x4 c0 = *(const f32x4*)(ap);
            f32x4 c1 = *(const f32x4*)(ap + 4);
            f32x4 c2 = *(const f32x4*)(ap + 8);
            f32x4 c3 = *(const f32x4*)(ap + 12);
#pragma unroll
            for (int i = 0; i < 4; ++i) {
                A.e[i]      = (__bf16)c0[i];
                A.e[4 + i]  = (__bf16)c1[i];
                A.e[8 + i]  = (__bf16)c2[i];
                A.e[12 + i] = (__bf16)c3[i];
            }
        } else {
#pragma unroll
            for (int i = 0; i < 16; ++i) A.e[i] = (__bf16)0.0f;
        }

        // ---- B fragments: contiguous 32B per lane from swizzled Wb --------
        const __bf16* bbase =
            Wb + (((size_t)kc * HDIM + nbase + l16) << 5) + (lhalf << 4);

#define SUBTILE(s, accv)                                                        \
        {                                                                       \
            union { v16bf v; u32x4 q[2]; } Bf;                                  \
            const u32x4* bq = (const u32x4*)(bbase + (s) * (16 * 32));          \
            Bf.q[0] = bq[0];                                                    \
            Bf.q[1] = bq[1];                                                    \
            accv = __builtin_amdgcn_wmma_f32_16x16x32_bf16(                     \
                false, A.v, false, Bf.v, (short)0, accv, false, false);         \
        }
        SUBTILE(0, acc0)
        SUBTILE(1, acc1)
        SUBTILE(2, acc2)
        SUBTILE(3, acc3)
#undef SUBTILE
    }

    // ---- relu + reduce over the 16 rows of this M-tile, accumulate --------
    // C layout: VGPR r, lane<16 -> (M=r, N=l16); lane>=16 -> (M=8+r, N=l16).
    // Sum of 8 regs per lane = partial col-sum over 8 rows; combine the two
    // lane-halves with a SWAPX16 ds_swizzle, then one atomic per column.
#define REDUCE(s, accv)                                                         \
    {                                                                           \
        float cs = 0.0f;                                                        \
        _Pragma("unroll")                                                       \
        for (int r = 0; r < 8; ++r) cs += fmaxf(accv[r], 0.0f);                 \
        cs += __int_as_float(                                                   \
            __builtin_amdgcn_ds_swizzle(__float_as_int(cs), 0x401F));           \
        if (lhalf == 0)                                                         \
            atomicAdd(&hs[b * HDIM + nbase + (s) * 16 + l16], cs);              \
    }
    REDUCE(0, acc0)
    REDUCE(1, acc1)
    REDUCE(2, acc2)
    REDUCE(3, acc3)
#undef REDUCE
}

// ---- final: out[b,o] = sum_h (h1s*h2s)[b,h] * Wp[h,o] + bp[o]*196^2 --------
__global__ void final_proj_kernel(const float* __restrict__ h1s, const float* __restrict__ h2s,
                                  const float* __restrict__ Wp,  const float* __restrict__ bp,
                                  float* __restrict__ out) {
    const int b = blockIdx.x;
    const int o = threadIdx.x;  // 128
    float a = 0.0f;
    for (int h = 0; h < HDIM; ++h) {
        float s = h1s[b * HDIM + h] * h2s[b * HDIM + h];
        a = fmaf(s, Wp[h * ODIM + o], a);
    }
    out[b * ODIM + o] = a + bp[o] * (float)(NROW * NROW);
}

// ---- launcher --------------------------------------------------------------
extern "C" void kernel_launch(void* const* d_in, const int* in_sizes, int n_in,
                              void* d_out, int out_size, void* d_ws, size_t ws_size,
                              hipStream_t stream) {
    const float* x1 = (const float*)d_in[0];
    const float* x2 = (const float*)d_in[1];
    const float* W1 = (const float*)d_in[2];
    const float* W2 = (const float*)d_in[3];
    const float* Wp = (const float*)d_in[4];
    const float* bp = (const float*)d_in[5];
    float* out = (float*)d_out;

    // workspace layout
    const size_t wbBytes = (size_t)CIN * HDIM * sizeof(__bf16);  // 786432 each
    char* ws = (char*)d_ws;
    __bf16* Wb1 = (__bf16*)(ws);
    __bf16* Wb2 = (__bf16*)(ws + wbBytes);
    float*  h1s = (float*)(ws + 2 * wbBytes);                    // [16,512]
    float*  h2s = (float*)(ws + 2 * wbBytes + (size_t)NB * HDIM * sizeof(float));

    // 1) zero both hsum buffers (contiguous: 2*16*512 floats)
    zero_f32_kernel<<<dim3((2 * NB * HDIM + 255) / 256), dim3(256), 0, stream>>>(
        h1s, 2 * NB * HDIM);

    // 2) swizzle + convert weights to bf16 B-fragment layout
    const int wElems = CIN * HDIM;
    swizzleW_kernel<<<dim3((wElems + 255) / 256), dim3(256), 0, stream>>>(W1, Wb1);
    swizzleW_kernel<<<dim3((wElems + 255) / 256), dim3(256), 0, stream>>>(W2, Wb2);

    // 3) fused WMMA GEMM + relu + row-sum for both inputs
    gemm_relu_colsum_kernel<<<dim3(MTILE, NB, 2), dim3(256), 0, stream>>>(
        x1, x2, Wb1, Wb2, h1s, h2s);

    // 4) tiny final projection
    final_proj_kernel<<<dim3(NB), dim3(ODIM), 0, stream>>>(h1s, h2s, Wp, bp, out);
}
